// FusionNet2_4105988735573
// MI455X (gfx1250) — matrix operand order repaired and checked by reading, not compiled, not run on hardware
//
#include <hip/hip_runtime.h>

#define NN 50000
#define NE 500000

typedef __bf16 bf16;
typedef __attribute__((ext_vector_type(16))) __bf16 bf16x16;
typedef __attribute__((ext_vector_type(8)))  __bf16 bf16x8;
typedef __attribute__((ext_vector_type(8)))  float  f32x8;
typedef unsigned int u32x4 __attribute__((ext_vector_type(4)));
typedef int          i32x4 __attribute__((ext_vector_type(4)));
typedef int          i32x8 __attribute__((ext_vector_type(8)));

struct P {
  const float* x; const int* ei; const float* ea;
  // packed bf16 weights (fragment-native layout)
  const bf16 *Wc1,*Wc2,*Wc3,*Wq,*Wk,*Wv,*Wih,*Whh,*Wself,*Wout;
  const bf16 *Wm0a,*Wm0b,*Wm1a,*Wm1b,*Waggr,*Wmlp1,*Wmlp2;
  // biases (f32, from inputs)
  const float *b_c1,*b_c2,*b_c3,*b_q,*b_k,*b_v,*b_ih,*b_hh,*b_self,*b_out;
  const float *b_m0a,*b_m0b,*b_m1a,*b_m1b,*b_aggr,*b_mlp1,*b_mlp2;
  // layernorm gamma (g_) / beta (e_)
  const float *g_ln1,*e_ln1,*g_ln2,*e_ln2,*g_ln3,*e_ln3;
  const float *g_n1,*e_n1,*g_n2,*e_n2;
  const float *g_m0,*e_m0,*g_m1,*e_m1,*g_a1,*e_a1,*g_a2,*e_a2;
  // workspace buffers
  float *cbuf,*cnbuf,*qnbuf,*mbuf,*sbuf,*aggv,*alphab;
  bf16* vb;
  float* out;
};

__device__ __forceinline__ int lane() { return threadIdx.x & 31; }

// wave-private LDS staging: only need DS-counter drain (lanes are lockstep in a wave)
__device__ __forceinline__ void lds_fence() { asm volatile("s_wait_dscnt 0" ::: "memory"); }

// ---- Tensor Data Mover: DMA nelem bf16 from global into LDS (1-D tile) ----
// D# group0: count=1, lds_addr, global_addr[56:0], type=2 ("image")
// D# group1: data_size=1 (2 bytes), tensor_dim0 = tile_dim0 = nelem, tensor_dim1 = 1
// This toolchain exposes the 6-arg builtin: (g0, g1, g2, g3, g_extra, cpol).
__device__ __forceinline__ void tdm_load(const bf16* gsrc, bf16* ldst, int nelem) {
  unsigned long long ga = (unsigned long long)(uintptr_t)gsrc;
  unsigned int la = (unsigned int)(uintptr_t)ldst;   // LDS aperture: low 32 bits = offset
  u32x4 g0;
  g0[0] = 1u;                                                     // count=1
  g0[1] = la;                                                     // lds_addr
  g0[2] = (unsigned int)ga;                                       // global_addr[31:0]
  g0[3] = ((unsigned int)(ga >> 32) & 0x01FFFFFFu) | (2u << 30);  // addr[56:32], type=2
  unsigned int td0 = (unsigned int)nelem;
  i32x8 g1;
  g1[0] = (int)(1u << 16);                       // wg_mask=0, data_size=1 (2B)
  g1[1] = (int)((td0 & 0xFFFFu) << 16);          // tensor_dim0[15:0]
  g1[2] = (int)((td0 >> 16) | (1u << 16));       // tensor_dim0[31:16], tensor_dim1=1
  g1[3] = (int)((td0 & 0xFFFFu) << 16);          // tensor_dim1[31:16]=0, tile_dim0
  g1[4] = 0;                                     // tile_dim1=0, tile_dim2=0
  g1[5] = (int)td0;                              // tensor_dim0_stride[31:0]
  g1[6] = 0;
  g1[7] = 0;
  i32x4 z4 = {0, 0, 0, 0};
  i32x8 z8 = {0, 0, 0, 0, 0, 0, 0, 0};
  __builtin_amdgcn_tensor_load_to_lds(g0, g1, z4, z4, z8, 0);
}

// A fragment (16x32 bf16) from LDS row-major [16 x ldk].
// lane l<16: row=l, K = kb+{0..7, 16..23}; lane>=16: row=l-16, K = kb+{8..15, 24..31}
__device__ __forceinline__ bf16x16 load_a(const bf16* s, int kb, int ldk) {
  int l = lane();
  const bf16* pp = s + (l & 15) * ldk + kb + ((l >> 4) << 3);
  bf16x8 lo = *(const bf16x8*)pp;
  bf16x8 hi = *(const bf16x8*)(pp + 16);
  bf16x16 a;
#pragma unroll
  for (int i = 0; i < 8; ++i) { a[i] = lo[i]; a[i + 8] = hi[i]; }
  return a;
}

// B fragment: packed so each lane does one contiguous 32-byte load (global or LDS)
__device__ __forceinline__ bf16x16 load_b(const bf16* wp, int ct, int kk, int k32) {
  const bf16* pp = wp + ((((size_t)ct * k32 + kk) * 32 + lane()) << 4);
  return *(const bf16x16*)pp;
}

__device__ __forceinline__ f32x8 wmma_bf16(bf16x16 a, bf16x16 b, f32x8 c) {
  return __builtin_amdgcn_wmma_f32_16x16x32_bf16(false, a, false, b, (short)0, c,
                                                 false, false);
}

// 16-row tile x (NT*16) cols GEMM, K = K32*32, bias-initialized accumulators
template<int NT, int K32>
__device__ __forceinline__ void gemm(const bf16* sA, const bf16* wp, const float* bias,
                                     int ctBase, f32x8* acc, bool accumulate) {
  int c0 = threadIdx.x & 15;
#pragma unroll
  for (int t = 0; t < NT; ++t) {
    int ct = ctBase + t;
    float bv = bias[ct * 16 + c0];
    f32x8 c;
#pragma unroll
    for (int r = 0; r < 8; ++r) c[r] = bv;
#pragma unroll
    for (int kk = 0; kk < K32; ++kk) {
      bf16x16 a = load_a(sA, kk * 32, K32 * 32);
      bf16x16 b = load_b(wp, ct, kk, K32);
      c = wmma_bf16(a, b, c);
    }
    if (accumulate) {
#pragma unroll
      for (int r = 0; r < 8; ++r) acc[t][r] += c[r];
    } else {
      acc[t] = c;
    }
  }
}

// In-register LayerNorm over 128 cols held as 8 C-fragments.
// C layout: reg r -> row (lane<16 ? r : r+8), col = 16*t + (lane&15).
__device__ __forceinline__ void layernorm8(f32x8* acc, const float* g, const float* b) {
  float mean[8], rstd[8];
#pragma unroll
  for (int r = 0; r < 8; ++r) {
    float s = 0.f, q = 0.f;
#pragma unroll
    for (int t = 0; t < 8; ++t) { float v = acc[t][r]; s += v; q += v * v; }
#pragma unroll
    for (int off = 1; off < 16; off <<= 1) {
      s += __shfl_xor(s, off, 16);
      q += __shfl_xor(q, off, 16);
    }
    float m = s * (1.f / 128.f);
    float var = q * (1.f / 128.f) - m * m;
    mean[r] = m; rstd[r] = rsqrtf(var + 1e-5f);
  }
  int c0 = threadIdx.x & 15;
#pragma unroll
  for (int t = 0; t < 8; ++t) {
    float gv = g[t * 16 + c0], bv = b[t * 16 + c0];
#pragma unroll
    for (int r = 0; r < 8; ++r) acc[t][r] = (acc[t][r] - mean[r]) * rstd[r] * gv + bv;
  }
}

__device__ __forceinline__ void relu8(f32x8* acc) {
#pragma unroll
  for (int t = 0; t < 8; ++t)
#pragma unroll
    for (int r = 0; r < 8; ++r) acc[t][r] = fmaxf(acc[t][r], 0.f);
}

template<int NT>
__device__ __forceinline__ void store_c_lds(const f32x8* acc, bf16* s, int ldk, int colBase) {
  int l = lane(); int rb = (l >> 4) << 3; int c0 = l & 15;
#pragma unroll
  for (int t = 0; t < NT; ++t)
#pragma unroll
    for (int r = 0; r < 8; ++r)
      s[(rb + r) * ldk + colBase + t * 16 + c0] = (bf16)acc[t][r];
}

__device__ __forceinline__ void store_c_global(const f32x8* acc, float* g) {
  int l = lane(); int rb = (l >> 4) << 3; int c0 = l & 15;
#pragma unroll
  for (int t = 0; t < 8; ++t)
#pragma unroll
    for (int r = 0; r < 8; ++r)
      g[(rb + r) * 128 + t * 16 + c0] = acc[t][r];
}

// stage 16 contiguous rows (f32) -> LDS bf16 [16x128], coalesced
__device__ __forceinline__ void stage_rows(const float* src, bf16* s) {
  int l = lane();
#pragma unroll
  for (int i = 0; i < 64; ++i) { int f = i * 32 + l; s[f] = (bf16)src[f]; }
}

// gather 16 rows via index array -> LDS bf16
__device__ __forceinline__ void stage_gather(const float* base, const int* idx, int ebase,
                                             bf16* s) {
  int l = lane();
#pragma unroll
  for (int i = 0; i < 64; ++i) {
    int f = i * 32 + l; int row = f >> 7; int col = f & 127;
    s[f] = (bf16)base[(size_t)idx[ebase + row] * 128 + col];
  }
}

__device__ __forceinline__ void atomicMaxF32(float* addr, float val) {
  unsigned int* ua = (unsigned int*)addr;
  unsigned int cur = *ua;
  while (__uint_as_float(cur) < val) {
    unsigned int prev = atomicCAS(ua, cur, __float_as_uint(val));
    if (prev == cur) break;
    cur = prev;
  }
}

// ---- pack f32 weight [K x Nout] into fragment-native bf16 layout ----
__global__ void k_pack(const float* w, bf16* wp, int K, int Nout) {
  int pidx = blockIdx.x * 256 + threadIdx.x;
  if (pidx >= K * Nout) return;
  int j = pidx & 15;
  int ln = (pidx >> 4) & 31;
  int rest = pidx >> 9;
  int k32 = K >> 5;
  int kk = rest % k32;
  int ct = rest / k32;
  int k = kk * 32 + ((ln >> 4) << 3) + (j & 7) + ((j >> 3) << 4);
  int n = ct * 16 + (ln & 15);
  wp[pidx] = (bf16)w[(size_t)k * Nout + n];
}

__global__ void k_init(float* aggv, float* mbuf, float* sbuf) {
  int i = blockIdx.x * 256 + threadIdx.x;
  if (i < NN * 128) aggv[i] = 0.f;
  if (i < NN * 8) { mbuf[i] = -3.0e38f; sbuf[i] = 0.f; }
}

// ---- node kernel: c = single_embed(x); cn = ln(norm1,c); q = lin_q(cn) ----
// Weights are TDM-staged into block-shared LDS once per layer (lockstep waves).
__global__ __launch_bounds__(128) void k_node(P p) {
  __shared__ alignas(64) bf16 sA[4][2048];
  __shared__ alignas(64) bf16 sB[4][2048];
  __shared__ alignas(64) bf16 wlds[16384];     // 32 KB shared weight buffer
  int w = threadIdx.x >> 5;
  int tile = blockIdx.x * 4 + w;
  if (tile >= NN / 16) tile = 0;   // tail waves duplicate tile 0 (idempotent writes)
  int nb = tile * 16;
  bf16* a = sA[w]; bf16* bb = sB[w];

  auto loadW = [&](const bf16* wsrc) {
    __syncthreads();                           // all waves done reading previous weights
    if (w == 0) {
      tdm_load(wsrc, wlds, 16384);             // async DMA, TENSORcnt++
      __builtin_amdgcn_s_wait_tensorcnt(0);    // drain before making visible
    }
    __syncthreads();                           // weights visible to all waves
  };

  stage_rows(p.x + (size_t)nb * 128, a);
  lds_fence();
  f32x8 acc[8];
  loadW(p.Wc1);
  gemm<8, 4>(a, wlds, p.b_c1, 0, acc, false);
  layernorm8(acc, p.g_ln1, p.e_ln1); relu8(acc);
  store_c_lds<8>(acc, bb, 128, 0); lds_fence();
  loadW(p.Wc2);
  gemm<8, 4>(bb, wlds, p.b_c2, 0, acc, false);
  layernorm8(acc, p.g_ln2, p.e_ln2); relu8(acc);
  store_c_lds<8>(acc, a, 128, 0); lds_fence();
  loadW(p.Wc3);
  gemm<8, 4>(a, wlds, p.b_c3, 0, acc, false);
  layernorm8(acc, p.g_ln3, p.e_ln3);
  store_c_global(acc, p.cbuf + (size_t)nb * 128);        // c
  layernorm8(acc, p.g_n1, p.e_n1);
  store_c_global(acc, p.cnbuf + (size_t)nb * 128);       // cn
  store_c_lds<8>(acc, bb, 128, 0); lds_fence();
  loadW(p.Wq);
  gemm<8, 4>(bb, wlds, p.b_q, 0, acc, false);
  store_c_global(acc, p.qnbuf + (size_t)nb * 128);       // q
}

// ---- edge kernel: nbr = multi_embed(x[src], ea); k,v; alpha; seg-max ----
__global__ __launch_bounds__(128) void k_edge(P p) {
  __shared__ alignas(64) bf16 s0[4][2048];
  __shared__ alignas(64) bf16 s1[4][2048];
  __shared__ alignas(64) bf16 s2[4][2048];
  int w = threadIdx.x >> 5;
  int tile = blockIdx.x * 4 + w;
  if (tile >= NE / 16) return;
  int eb = tile * 16;
  bf16* sx = s0[w]; bf16* se = s1[w]; bf16* sw = s2[w];
  stage_gather(p.x, p.ei, eb, sx);                 // x[src]
  stage_rows(p.ea + (size_t)eb * 128, se);         // edge_attr
  lds_fence();
  f32x8 acc[8], u[8];
  // branch m0 on x[src]
  gemm<8, 4>(sx, p.Wm0a, p.b_m0a, 0, acc, false);
  layernorm8(acc, p.g_m0, p.e_m0); relu8(acc);
  store_c_lds<8>(acc, sw, 128, 0); lds_fence();
  gemm<8, 4>(sw, p.Wm0b, p.b_m0b, 0, u, false);
  // branch m1 on edge_attr, accumulate
  gemm<8, 4>(se, p.Wm1a, p.b_m1a, 0, acc, false);
  layernorm8(acc, p.g_m1, p.e_m1); relu8(acc);
  store_c_lds<8>(acc, sw, 128, 0); lds_fence();
  gemm<8, 4>(sw, p.Wm1b, p.b_m1b, 0, u, true);
  // aggregate head
  layernorm8(u, p.g_a1, p.e_a1); relu8(u);
  store_c_lds<8>(u, sw, 128, 0); lds_fence();
  gemm<8, 4>(sw, p.Waggr, p.b_aggr, 0, acc, false);
  layernorm8(acc, p.g_a2, p.e_a2);                 // nbr
  store_c_lds<8>(acc, sw, 128, 0); lds_fence();
  gemm<8, 4>(sw, p.Wk, p.b_k, 0, acc, false);      // k
  gemm<8, 4>(sw, p.Wv, p.b_v, 0, u, false);        // v
  int l = lane(); int rb = (l >> 4) << 3; int c0 = l & 15;
#pragma unroll
  for (int t = 0; t < 8; ++t)
#pragma unroll
    for (int r = 0; r < 8; ++r)
      p.vb[(size_t)(eb + rb + r) * 128 + t * 16 + c0] = (bf16)u[t][r];
  int drow[8];
#pragma unroll
  for (int r = 0; r < 8; ++r) drow[r] = p.ei[NE + eb + rb + r];   // dst
  // alpha[e,h] = (q[dst] . k)/4 ; head h == col-tile t
#pragma unroll
  for (int t = 0; t < 8; ++t) {
    float q[8];
#pragma unroll
    for (int r = 0; r < 8; ++r)
      q[r] = p.qnbuf[(size_t)drow[r] * 128 + t * 16 + c0] * acc[t][r];
#pragma unroll
    for (int off = 1; off < 16; off <<= 1)
#pragma unroll
      for (int r = 0; r < 8; ++r) q[r] += __shfl_xor(q[r], off, 16);
    if (c0 == t) {
#pragma unroll
      for (int r = 0; r < 8; ++r) {
        float av = q[r] * 0.25f;    // 1/sqrt(16)
        int e = eb + rb + r;
        p.alphab[(size_t)e * 8 + t] = av;
        atomicMaxF32(p.mbuf + (size_t)drow[r] * 8 + t, av);
      }
    }
  }
}

// ---- scatter: unnormalized softmax accumulation ----
__global__ __launch_bounds__(256) void k_scatter(P p) {
  size_t i = (size_t)blockIdx.x * 256 + threadIdx.x;
  if (i >= (size_t)NE * 128) return;
  int e = (int)(i >> 7); int col = (int)(i & 127); int h = col >> 4;
  int d = p.ei[NE + e];
  float a = __expf(p.alphab[(size_t)e * 8 + h] - p.mbuf[(size_t)d * 8 + h]);
  atomicAdd(p.aggv + (size_t)d * 128 + col, a * (float)p.vb[i]);
  if ((col & 15) == 0) atomicAdd(p.sbuf + (size_t)d * 8 + h, a);
}

// ---- finalize: gate, out_proj, norm2, MLP, residuals ----
__global__ __launch_bounds__(64) void k_final(P p) {
  __shared__ alignas(64) bf16 sA[2][2048];
  __shared__ alignas(64) bf16 sB[2][2048];
  __shared__ alignas(64) bf16 sH[2][8192];
  int w = threadIdx.x >> 5;
  int tile = blockIdx.x * 2 + w;
  if (tile >= NN / 16) return;
  int nb = tile * 16;
  bf16* a = sA[w]; bf16* b = sB[w]; bf16* h2 = sH[w];
  int l = lane(); int rb = (l >> 4) << 3; int c0 = l & 15;
  f32x8 agg[8];
#pragma unroll
  for (int t = 0; t < 8; ++t)
#pragma unroll
    for (int r = 0; r < 8; ++r) {
      int row = nb + rb + r; int col = t * 16 + c0;
      float sv = p.sbuf[(size_t)row * 8 + t];
      agg[t][r] = (sv > 0.f) ? p.aggv[(size_t)row * 128 + col] / sv : 0.f;
    }
  store_c_lds<8>(agg, a, 128, 0);
  stage_rows(p.cnbuf + (size_t)nb * 128, b);
  lds_fence();
  f32x8 g8[8], s8[8];
  gemm<8, 4>(a, p.Wih, p.b_ih, 0, g8, false);
  gemm<8, 4>(b, p.Whh, p.b_hh, 0, g8, true);
  gemm<8, 4>(b, p.Wself, p.b_self, 0, s8, false);
  f32x8 upd[8];
#pragma unroll
  for (int t = 0; t < 8; ++t)
#pragma unroll
    for (int r = 0; r < 8; ++r) {
      float gate = 1.f / (1.f + __expf(-g8[t][r]));
      upd[t][r] = agg[t][r] + gate * (s8[t][r] - agg[t][r]);
    }
  store_c_lds<8>(upd, a, 128, 0); lds_fence();
  f32x8 o[8];
  gemm<8, 4>(a, p.Wout, p.b_out, 0, o, false);
#pragma unroll
  for (int t = 0; t < 8; ++t)
#pragma unroll
    for (int r = 0; r < 8; ++r)
      o[t][r] += p.cbuf[(size_t)(nb + rb + r) * 128 + t * 16 + c0];   // out = c + proj
  f32x8 hh[8];
#pragma unroll
  for (int t = 0; t < 8; ++t) hh[t] = o[t];
  layernorm8(hh, p.g_n2, p.e_n2);
  store_c_lds<8>(hh, a, 128, 0); lds_fence();
#pragma unroll
  for (int ch = 0; ch < 4; ++ch) {       // D -> 4D in 128-col chunks
    f32x8 m1[8];
    gemm<8, 4>(a, p.Wmlp1, p.b_mlp1, ch * 8, m1, false);
    relu8(m1);
    store_c_lds<8>(m1, h2, 512, ch * 128);
  }
  lds_fence();
  f32x8 m2[8];
  gemm<8, 16>(h2, p.Wmlp2, p.b_mlp2, 0, m2, false);   // 4D -> D
#pragma unroll
  for (int t = 0; t < 8; ++t)
#pragma unroll
    for (int r = 0; r < 8; ++r)
      p.out[(size_t)(nb + rb + r) * 128 + t * 16 + c0] = o[t][r] + m2[t][r];
}

extern "C" void kernel_launch(void* const* d_in, const int* in_sizes, int n_in,
                              void* d_out, int out_size, void* d_ws, size_t ws_size,
                              hipStream_t stream) {
  (void)in_sizes; (void)n_in; (void)out_size; (void)ws_size;
  char* ws = (char*)d_ws;
  size_t off = 0;
  auto take = [&](size_t bytes) -> char* {
    char* r = ws + off; off = (off + bytes + 255) & ~(size_t)255; return r;
  };

  P p{};
  p.x  = (const float*)d_in[0];
  p.ei = (const int*)d_in[1];
  p.ea = (const float*)d_in[2];
  // params flattened: top-level dict insertion order, pytree dicts sorted by key
  p.b_c1 = (const float*)d_in[3];   /* center.lin1.{b,w}=3,4 */
  p.b_c2 = (const float*)d_in[5];
  p.b_c3 = (const float*)d_in[7];
  p.e_ln1 = (const float*)d_in[9];  p.g_ln1 = (const float*)d_in[10];
  p.e_ln2 = (const float*)d_in[11]; p.g_ln2 = (const float*)d_in[12];
  p.e_ln3 = (const float*)d_in[13]; p.g_ln3 = (const float*)d_in[14];
  p.b_hh   = (const float*)d_in[15];
  p.b_ih   = (const float*)d_in[17];
  p.b_k    = (const float*)d_in[19];
  p.b_q    = (const float*)d_in[21];
  p.b_self = (const float*)d_in[23];
  p.b_v    = (const float*)d_in[25];
  p.b_mlp1 = (const float*)d_in[27];
  p.b_mlp2 = (const float*)d_in[29];
  p.b_aggr = (const float*)d_in[31];
  p.e_a1 = (const float*)d_in[33]; p.g_a1 = (const float*)d_in[34];
  p.e_a2 = (const float*)d_in[35]; p.g_a2 = (const float*)d_in[36];
  p.b_m0a = (const float*)d_in[37];
  p.b_m0b = (const float*)d_in[39];
  p.e_m0 = (const float*)d_in[41]; p.g_m0 = (const float*)d_in[42];
  p.b_m1a = (const float*)d_in[43];
  p.b_m1b = (const float*)d_in[45];
  p.e_m1 = (const float*)d_in[47]; p.g_m1 = (const float*)d_in[48];
  p.e_n1 = (const float*)d_in[49]; p.g_n1 = (const float*)d_in[50];
  p.e_n2 = (const float*)d_in[51]; p.g_n2 = (const float*)d_in[52];
  p.b_out = (const float*)d_in[53];

  // packed bf16 weights
  bf16* Wc1   = (bf16*)take(16384 * 2);
  bf16* Wc2   = (bf16*)take(16384 * 2);
  bf16* Wc3   = (bf16*)take(16384 * 2);
  bf16* Wq    = (bf16*)take(16384 * 2);
  bf16* Wk    = (bf16*)take(16384 * 2);
  bf16* Wv    = (bf16*)take(16384 * 2);
  bf16* Wih   = (bf16*)take(16384 * 2);
  bf16* Whh   = (bf16*)take(16384 * 2);
  bf16* Wself = (bf16*)take(16384 * 2);
  bf16* Wout  = (bf16*)take(16384 * 2);
  bf16* Wm0a  = (bf16*)take(16384 * 2);
  bf16* Wm0b  = (bf16*)take(16384 * 2);
  bf16* Wm1a  = (bf16*)take(16384 * 2);
  bf16* Wm1b  = (bf16*)take(16384 * 2);
  bf16* Waggr = (bf16*)take(16384 * 2);
  bf16* Wmlp1 = (bf16*)take(65536 * 2);
  bf16* Wmlp2 = (bf16*)take(65536 * 2);
  p.Wc1 = Wc1; p.Wc2 = Wc2; p.Wc3 = Wc3; p.Wq = Wq; p.Wk = Wk; p.Wv = Wv;
  p.Wih = Wih; p.Whh = Whh; p.Wself = Wself; p.Wout = Wout;
  p.Wm0a = Wm0a; p.Wm0b = Wm0b; p.Wm1a = Wm1a; p.Wm1b = Wm1b; p.Waggr = Waggr;
  p.Wmlp1 = Wmlp1; p.Wmlp2 = Wmlp2;

  p.cbuf   = (float*)take((size_t)NN * 128 * 4);
  p.cnbuf  = (float*)take((size_t)NN * 128 * 4);
  p.qnbuf  = (float*)take((size_t)NN * 128 * 4);
  p.mbuf   = (float*)take((size_t)NN * 8 * 4);
  p.sbuf   = (float*)take((size_t)NN * 8 * 4);
  p.aggv   = (float*)take((size_t)NN * 128 * 4);
  p.alphab = (float*)take((size_t)NE * 8 * 4);
  p.vb     = (bf16*)take((size_t)NE * 128 * 2);
  p.out = (float*)d_out;

  auto pack = [&](const void* src, bf16* dst, int K, int Nout) {
    int total = K * Nout;
    hipLaunchKernelGGL(k_pack, dim3((total + 255) / 256), dim3(256), 0, stream,
                       (const float*)src, dst, K, Nout);
  };
  pack(d_in[4],  Wc1, 128, 128);
  pack(d_in[6],  Wc2, 128, 128);
  pack(d_in[8],  Wc3, 128, 128);
  pack(d_in[16], Whh, 128, 128);
  pack(d_in[18], Wih, 128, 128);
  pack(d_in[20], Wk,  128, 128);
  pack(d_in[22], Wq,  128, 128);
  pack(d_in[24], Wself, 128, 128);
  pack(d_in[26], Wv,  128, 128);
  pack(d_in[28], Wmlp1, 128, 512);
  pack(d_in[30], Wmlp2, 512, 128);
  pack(d_in[32], Waggr, 128, 128);
  pack(d_in[38], Wm0a, 128, 128);
  pack(d_in[40], Wm0b, 128, 128);
  pack(d_in[44], Wm1a, 128, 128);
  pack(d_in[46], Wm1b, 128, 128);
  pack(d_in[54], Wout, 128, 128);

  hipLaunchKernelGGL(k_init, dim3((NN * 128 + 255) / 256), dim3(256), 0, stream,
                     p.aggv, p.mbuf, p.sbuf);
  hipLaunchKernelGGL(k_node, dim3((NN / 16 + 3) / 4), dim3(128), 0, stream, p);
  hipLaunchKernelGGL(k_edge, dim3((NE / 16 + 3) / 4), dim3(128), 0, stream, p);
  hipLaunchKernelGGL(k_scatter, dim3((int)(((size_t)NE * 128 + 255) / 256)), dim3(256),
                     0, stream, p);
  hipLaunchKernelGGL(k_final, dim3((NN / 16 + 1) / 2), dim3(64), 0, stream, p);
}
